// CriticRDDPG_72456098284042
// MI455X (gfx1250) — compile-verified, matrix operand order checked by reading, not run on hardware
//
#include <hip/hip_runtime.h>
#include <hip/hip_bf16.h>
#include <float.h>

// ---------------------------------------------------------------------------
// CriticRDDPG forward for MI455X (gfx1250), wave32 + WMMA bf16.
//   B=128, T=512, A=256, U_GRU=512, K_x=768, N_gates=1536
// Phase 1: weight prep + activations + h0 (tiny).
// Phase 2: wave-tiled bf16 WMMA GEMMs for inp2/inp3/x_proj (154 GFLOP bulk).
// Phase 3: persistent GRU scan kernel — single launch, 32 blocks, gru_rk slice
//          cached in LDS (padded rows: 1040B stride -> conflict-free 4-bank
//          rotation per lane) for all 512 steps; device-wide atomic barrier
//          between steps (replaces 512 kernel launches).
// Workspace ~520 MB.
// ---------------------------------------------------------------------------

typedef __attribute__((ext_vector_type(16))) __bf16 v16bf;
typedef __attribute__((ext_vector_type(8)))  float  v8f;

union BF16Tile { v16bf v; unsigned int u[8]; };

#define LDS_ROWW 520   // padded row stride in halfwords (1040 B; 1040%256=16 -> 4-bank rotate)

__device__ __forceinline__ unsigned short f2bf(float f) {
  union { float f; unsigned int u; } c; c.f = f;
  unsigned int u = c.u;
  u += 0x7FFFu + ((u >> 16) & 1u);          // round-to-nearest-even
  return (unsigned short)(u >> 16);
}

__device__ __forceinline__ float elu_f(float x) {
  return x > 0.0f ? x : (__expf(x) - 1.0f);
}
__device__ __forceinline__ float sigmoid_f(float x) {
  return 1.0f / (1.0f + __expf(-x));
}

// dword d of the per-lane v16bf tile covers K = kmap(d), kmap(d)+1  (+ half*8)
__device__ __forceinline__ int kmap(int d) { return ((d & 3) * 2) + ((d >> 2) * 16); }

// ---------------------------------------------------------------------------
// Setup kernels
// ---------------------------------------------------------------------------

// in: f32 [R,C] row-major  ->  out: bf16 [C,R] row-major (transposed)
__global__ void k_transpose_bf16(const float* __restrict__ in,
                                 unsigned short* __restrict__ out,
                                 int R, int C) {
  int idx = blockIdx.x * blockDim.x + threadIdx.x;
  if (idx >= R * C) return;
  int r = idx / C, c = idx % C;
  out[(size_t)c * R + r] = f2bf(in[idx]);
}

// x[:,0:256] = bf16(action*mu+mean),  m = b*T + t
__global__ void k_act(const float* __restrict__ action,
                      const float* __restrict__ mu,
                      const float* __restrict__ mean,
                      unsigned short* __restrict__ x) {
  int idx = blockIdx.x * blockDim.x + threadIdx.x;       // 65536*256
  int m = idx >> 8, a = idx & 255;
  int b = m >> 9;                                         // m / 512
  float v = action[(size_t)idx] * mu[b * 256 + a] + mean[b * 256 + a];
  x[(size_t)m * 768 + a] = f2bf(v);
}

// osc_bf16[m][k] = bf16(osc[b][t][k]), k < 64
__global__ void k_oscslice(const float* __restrict__ osc,
                           unsigned short* __restrict__ oscb) {
  int idx = blockIdx.x * blockDim.x + threadIdx.x;       // 65536*64
  int m = idx >> 6, k = idx & 63;
  oscb[(size_t)idx] = f2bf(osc[(size_t)m * 128 + k]);
}

// msrs[b][0:256]=elu(mot@W_mot+b) ; msrs[b][256:512]=elu(rob@W_rob+b)
__global__ void k_msrs(const float* __restrict__ mot, const float* __restrict__ rob,
                       const float* __restrict__ Wm, const float* __restrict__ bm,
                       const float* __restrict__ Wr, const float* __restrict__ br,
                       float* __restrict__ msrs) {
  int idx = blockIdx.x * blockDim.x + threadIdx.x;       // 128*512
  int b = idx >> 9, j = idx & 511;
  const float* inp; const float* W; const float* bb; int col;
  if (j < 256) { inp = mot + b * 64; W = Wm; bb = bm; col = j; }
  else         { inp = rob + b * 64; W = Wr; bb = br; col = j - 256; }
  float s = bb[col];
  for (int k = 0; k < 64; ++k) s += inp[k] * W[k * 256 + col];
  msrs[idx] = elu_f(s);
}

// h0 = elu(concat(ms,rs,state) @ W_comb + b_comb)
__global__ void k_h0(const float* __restrict__ msrs, const float* __restrict__ state,
                     const float* __restrict__ Wc, const float* __restrict__ bc,
                     float* __restrict__ h, unsigned short* __restrict__ hbf) {
  int idx = blockIdx.x * blockDim.x + threadIdx.x;       // 128*512
  int b = idx >> 9, u = idx & 511;
  float s = bc[u];
  for (int k = 0; k < 512; ++k)  s += msrs[b * 512 + k]  * Wc[(size_t)k * 512 + u];
  for (int k = 0; k < 512; ++k)  s += state[b * 512 + k] * Wc[(size_t)(512 + k) * 512 + u];
  float v = elu_f(s);
  h[idx] = v;
  hbf[idx] = f2bf(v);
}

__global__ void k_init(float* __restrict__ out_max, unsigned int* __restrict__ bar) {
  int idx = blockIdx.x * blockDim.x + threadIdx.x;       // 128*512
  out_max[idx] = -FLT_MAX;
  if (idx == 0) *bar = 0u;
}

// ---------------------------------------------------------------------------
// Generic wave-tiled bf16 WMMA GEMM: C[M,N] = act(A[M,K] @ Bt[N,K]^T + bias)
// Each wave computes a 16x64 strip (4 accumulators, A reused).
// ---------------------------------------------------------------------------
template <bool OUT_BF16, bool DO_ELU>
__global__ void k_gemm(const unsigned short* __restrict__ A, int lda,
                       const unsigned short* __restrict__ Bt, int K,
                       const float* __restrict__ bias,
                       void* __restrict__ Cout, int ldc,
                       int Mtiles, int Ngroups) {
  int wave = (blockIdx.x * blockDim.x + threadIdx.x) >> 5;
  int lane = threadIdx.x & 31;
  if (wave >= Mtiles * Ngroups) return;                  // wave-uniform branch
  int mt = wave / Ngroups, ng = wave % Ngroups;
  int m0 = mt * 16, n0g = ng * 64;
  int l15 = lane & 15, half = lane >> 4;

  v8f acc[4];
  #pragma unroll
  for (int j = 0; j < 4; ++j)
    acc[j] = (v8f){0.f, 0.f, 0.f, 0.f, 0.f, 0.f, 0.f, 0.f};

  for (int kc = 0; kc < K; kc += 32) {
    BF16Tile a;
    const unsigned short* Ar = A + (size_t)(m0 + l15) * lda + kc + half * 8;
    #pragma unroll
    for (int d = 0; d < 8; ++d)
      a.u[d] = *(const unsigned int*)(Ar + kmap(d));
    #pragma unroll
    for (int j = 0; j < 4; ++j) {
      BF16Tile bm;
      const unsigned short* Br = Bt + (size_t)(n0g + j * 16 + l15) * K + kc + half * 8;
      #pragma unroll
      for (int d = 0; d < 8; ++d)
        bm.u[d] = *(const unsigned int*)(Br + kmap(d));
      acc[j] = __builtin_amdgcn_wmma_f32_16x16x32_bf16(
          false, a.v, false, bm.v, (short)0, acc[j], false, false);
    }
  }

  #pragma unroll
  for (int j = 0; j < 4; ++j) {
    int n = n0g + j * 16 + l15;
    float bv = bias ? bias[n] : 0.0f;
    #pragma unroll
    for (int r = 0; r < 8; ++r) {
      int m = m0 + r + 8 * half;
      float v = acc[j][r] + bv;
      if (DO_ELU) v = elu_f(v);
      if (OUT_BF16) ((unsigned short*)Cout)[(size_t)m * ldc + n] = f2bf(v);
      else          ((float*)Cout)[(size_t)m * ldc + n] = v;
    }
  }
}

// ---------------------------------------------------------------------------
// Persistent GRU scan. Single launch, 32 blocks x 256 threads (8 waves).
// Block nt owns output columns [nt*16, nt*16+16) of U_GRU for all 3 gates;
// its gru_rk slice (48 rows x 512 K bf16, padded to 1040B/row = ~49KB) is
// staged into LDS once and reused for all 512 timesteps. Wave w owns batch
// rows [w*16, w*16+16). Steps separated by a device-wide atomic barrier.
// ---------------------------------------------------------------------------
__global__ void k_gru_scan(float* h_a, unsigned short* hbf_a,
                           float* h_b, unsigned short* hbf_b,
                           const float* __restrict__ x_proj,
                           const unsigned short* __restrict__ rk_t, // [1536,512] bf16
                           const float* __restrict__ b_rec,         // [1536]
                           float* __restrict__ out_max,
                           unsigned int* bar) {
  __shared__ unsigned short lds_rk[48 * LDS_ROWW];       // 49,920 bytes

  const int nt   = blockIdx.x;                           // 0..31
  const int n0   = nt * 16;
  const int tid  = threadIdx.x;
  const int wave = tid >> 5;                             // mt 0..7
  const int lane = tid & 31;
  const int l15  = lane & 15, half = lane >> 4;
  const int m0   = wave * 16;

  // ---- stage this block's rk slice into LDS (rows: gate*16 + nl, cols K) ----
  {
    unsigned int* dst = (unsigned int*)lds_rk;
    for (int i = tid; i < 48 * 256; i += 256) {          // 256 payload dwords per row
      int r = i >> 8, dw = i & 255;
      int g = r >> 4, nl = r & 15;
      const unsigned int* s =
          (const unsigned int*)(rk_t + (size_t)(g * 512 + n0 + nl) * 512);
      dst[r * (LDS_ROWW / 2) + dw] = s[dw];              // padded row stride
    }
  }
  __syncthreads();

  float*          hi  = h_a;  unsigned short* hbi = hbf_a;
  float*          ho  = h_b;  unsigned short* hbo = hbf_b;

  const int   n   = n0 + l15;
  const float bzb = b_rec[n], brb = b_rec[512 + n], bhb = b_rec[1024 + n];
  const unsigned int* lds32 = (const unsigned int*)lds_rk;

  for (int t = 0; t < 512; ++t) {
    v8f accZ = (v8f){0.f,0.f,0.f,0.f,0.f,0.f,0.f,0.f};
    v8f accR = accZ, accH = accZ;

    for (int kc = 0; kc < 512; kc += 32) {
      BF16Tile a;
      const unsigned short* Ar = hbi + (size_t)(m0 + l15) * 512 + kc + half * 8;
      #pragma unroll
      for (int d = 0; d < 8; ++d)
        a.u[d] = *(const unsigned int*)(Ar + kmap(d));

      BF16Tile bz, br, bh;
      // dword index into padded LDS: row l15 (per-gate), cols kc + half*8 + kmap
      int base = (l15 * LDS_ROWW + kc + half * 8) >> 1;
      #pragma unroll
      for (int d = 0; d < 8; ++d) {
        int o = kmap(d) >> 1;
        bz.u[d] = lds32[base + o];                             // gate z rows  0..15
        br.u[d] = lds32[base + (16 * LDS_ROWW >> 1) + o];      // gate r rows 16..31
        bh.u[d] = lds32[base + (32 * LDS_ROWW >> 1) + o];      // gate h rows 32..47
      }
      accZ = __builtin_amdgcn_wmma_f32_16x16x32_bf16(false, a.v, false, bz.v, (short)0, accZ, false, false);
      accR = __builtin_amdgcn_wmma_f32_16x16x32_bf16(false, a.v, false, br.v, (short)0, accR, false, false);
      accH = __builtin_amdgcn_wmma_f32_16x16x32_bf16(false, a.v, false, bh.v, (short)0, accH, false, false);
    }

    #pragma unroll
    for (int r = 0; r < 8; ++r) {
      int m = m0 + r + 8 * half;                         // batch row
      size_t xbase = ((size_t)m * 512 + (size_t)t) * 1536;
      float xz = x_proj[xbase + n];
      float xr = x_proj[xbase + 512 + n];
      float xh = x_proj[xbase + 1024 + n];
      float hz = accZ[r] + bzb, hr = accR[r] + brb, hh = accH[r] + bhb;
      float z  = sigmoid_f(xz + hz);
      float rg = sigmoid_f(xr + hr);
      float hc = tanhf(xh + rg * hh);
      size_t hidx = (size_t)m * 512 + n;
      float hn = z * hi[hidx] + (1.0f - z) * hc;
      ho[hidx]  = hn;
      hbo[hidx] = f2bf(hn);
      out_max[hidx] = fmaxf(out_max[hidx], hn);
    }

    // ---- device-wide barrier: monotone counter, target 32*(t+1) ----
    __syncthreads();
    __threadfence();                                     // release this step's h writes
    if (tid == 0) {
      __hip_atomic_fetch_add(bar, 1u, __ATOMIC_ACQ_REL, __HIP_MEMORY_SCOPE_AGENT);
      unsigned int target = 32u * (unsigned int)(t + 1);
      while (__hip_atomic_load(bar, __ATOMIC_ACQUIRE, __HIP_MEMORY_SCOPE_AGENT) < target)
        __builtin_amdgcn_s_sleep(1);
    }
    __syncthreads();
    __threadfence();                                     // acquire other blocks' writes

    // swap ping-pong
    float* tf = hi; hi = ho; ho = tf;
    unsigned short* ts = hbi; hbi = hbo; hbo = ts;
  }
  // 512 steps: final h ends in h_a (written at t=511)
}

// out[0:128] = elu(max_h @ W_out + b_out) ; out[128:] = h_final
__global__ void k_final(const float* __restrict__ out_max,
                        const float* __restrict__ h_final,
                        const float* __restrict__ W_out,
                        const float* __restrict__ b_out,
                        float* __restrict__ d_out) {
  int idx = blockIdx.x * blockDim.x + threadIdx.x;
  if (idx < 128) {
    float s = b_out[0];
    for (int k = 0; k < 512; ++k) s += out_max[idx * 512 + k] * W_out[k];
    d_out[idx] = elu_f(s);
  } else if (idx < 128 + 128 * 512) {
    d_out[idx] = h_final[idx - 128];
  }
}

// ---------------------------------------------------------------------------
extern "C" void kernel_launch(void* const* d_in, const int* in_sizes, int n_in,
                              void* d_out, int out_size, void* d_ws, size_t ws_size,
                              hipStream_t stream) {
  (void)in_sizes; (void)n_in; (void)out_size; (void)ws_size;
  const float* motion = (const float*)d_in[0];
  const float* robot  = (const float*)d_in[1];
  // d_in[2] = osc_state : unused by the reference forward
  const float* action = (const float*)d_in[3];
  const float* osc    = (const float*)d_in[4];
  const float* mu     = (const float*)d_in[5];
  const float* mean   = (const float*)d_in[6];
  const float* state  = (const float*)d_in[7];
  const float* W_mot  = (const float*)d_in[8];
  const float* b_mot  = (const float*)d_in[9];
  const float* W_rob  = (const float*)d_in[10];
  const float* b_rob  = (const float*)d_in[11];
  const float* W_comb = (const float*)d_in[12];
  const float* b_comb = (const float*)d_in[13];
  const float* W_oscr = (const float*)d_in[14];
  const float* b_oscr = (const float*)d_in[15];
  const float* W_osci = (const float*)d_in[16];
  const float* b_osci = (const float*)d_in[17];
  const float* gru_k  = (const float*)d_in[18];
  const float* gru_rk = (const float*)d_in[19];
  const float* gru_b  = (const float*)d_in[20];   // [2,1536]
  const float* W_out  = (const float*)d_in[21];
  const float* b_out  = (const float*)d_in[22];
  float* out = (float*)d_out;

  const size_t M = 65536;  // B*T

  size_t off = 0;
  auto carve = [&](size_t bytes) -> char* {
    char* p = (char*)d_ws + off;
    off += (bytes + 255) & ~(size_t)255;
    return p;
  };
  unsigned short* x_bf    = (unsigned short*)carve(M * 768 * 2);
  unsigned short* osc_bf  = (unsigned short*)carve(M * 64 * 2);
  float*          x_proj  = (float*)carve(M * 1536 * 4);
  unsigned short* rk_t    = (unsigned short*)carve((size_t)1536 * 512 * 2);
  unsigned short* gk_t    = (unsigned short*)carve((size_t)1536 * 768 * 2);
  unsigned short* woscr_t = (unsigned short*)carve((size_t)256 * 64 * 2);
  unsigned short* wosci_t = (unsigned short*)carve((size_t)256 * 192 * 2);
  float*          msrs    = (float*)carve(128 * 512 * 4);
  float*          h_a     = (float*)carve(128 * 512 * 4);
  float*          h_b     = (float*)carve(128 * 512 * 4);
  unsigned short* hbf_a   = (unsigned short*)carve(128 * 512 * 2);
  unsigned short* hbf_b   = (unsigned short*)carve(128 * 512 * 2);
  float*          omax    = (float*)carve(128 * 512 * 4);
  unsigned int*   bar     = (unsigned int*)carve(256);

  // ---- setup: weight transposes to bf16 [N,K] ----
  k_transpose_bf16<<<(64 * 256 + 255) / 256, 256, 0, stream>>>(W_oscr, woscr_t, 64, 256);
  k_transpose_bf16<<<(192 * 256 + 255) / 256, 256, 0, stream>>>(W_osci, wosci_t, 192, 256);
  k_transpose_bf16<<<(768 * 1536 + 255) / 256, 256, 0, stream>>>(gru_k, gk_t, 768, 1536);
  k_transpose_bf16<<<(512 * 1536 + 255) / 256, 256, 0, stream>>>(gru_rk, rk_t, 512, 1536);

  // ---- elementwise prep ----
  k_act<<<(int)(M * 256 / 256), 256, 0, stream>>>(action, mu, mean, x_bf);
  k_oscslice<<<(int)(M * 64 / 256), 256, 0, stream>>>(osc, osc_bf);
  k_msrs<<<(128 * 512) / 256, 256, 0, stream>>>(motion, robot, W_mot, b_mot, W_rob, b_rob, msrs);
  k_h0<<<(128 * 512) / 256, 256, 0, stream>>>(msrs, state, W_comb, b_comb, h_a, hbf_a);
  k_init<<<(128 * 512) / 256, 256, 0, stream>>>(omax, bar);

  // ---- inp2 = elu(osc[...,:64] @ W_oscr + b)  ->  x[:,256:512]  (bf16) ----
  {
    int Mt = (int)(M / 16), Ng = 256 / 64;
    int nwaves = Mt * Ng;
    k_gemm<true, true><<<(nwaves + 7) / 8, 256, 0, stream>>>(
        osc_bf, 64, woscr_t, 64, b_oscr, x_bf + 256, 768, Mt, Ng);
  }
  // ---- inp3 = elu(inp2[...,64:] @ W_osci + b) -> x[:,512:768]  (bf16) ----
  {
    int Mt = (int)(M / 16), Ng = 256 / 64;
    int nwaves = Mt * Ng;
    k_gemm<true, true><<<(nwaves + 7) / 8, 256, 0, stream>>>(
        x_bf + 320, 768, wosci_t, 192, b_osci, x_bf + 512, 768, Mt, Ng);
  }
  // ---- x_proj = x @ gru_k + gru_b[0]  (f32 [B,T,1536]) ----
  {
    int Mt = (int)(M / 16), Ng = 1536 / 64;
    int nwaves = Mt * Ng;
    k_gemm<false, false><<<(nwaves + 7) / 8, 256, 0, stream>>>(
        x_bf, 768, gk_t, 768, gru_b, x_proj, 1536, Mt, Ng);
  }

  // ---- persistent GRU scan: one launch, internal device-wide barriers ----
  k_gru_scan<<<32, 256, 0, stream>>>(h_a, hbf_a, h_b, hbf_b,
                                     x_proj, rk_t, gru_b + 1536, omax, bar);

  // after 512 steps (even), final h lives in h_a
  k_final<<<(128 + 128 * 512 + 255) / 256, 256, 0, stream>>>(omax, h_a, W_out, b_out, out);
}
